// DownSampleToken_13159779795141
// MI455X (gfx1250) — matrix operand order repaired and checked by reading, not compiled, not run on hardware
//
#include <hip/hip_runtime.h>
#include <hip/hip_bf16.h>
#include <cstdint>

// ---------------------------------------------------------------------------
// DownSampleToken pipeline for MI455X (gfx1250, wave32, WMMA + TDM).
// All GEMMs use V_WMMA_F32_16X16X32_BF16 with split-bf16 (hi+lo) compensation
// (3 WMMAs per tile) for ~f32 accuracy at bf16-WMMA rates. The xds GEMM
// stages its shared B tiles through LDS with the Tensor Data Mover
// (tensor_load_to_lds, TENSORcnt), double-buffered to overlap DMA and WMMA.
// ---------------------------------------------------------------------------

#define DI __device__ __forceinline__

constexpr int B_   = 4;
constexpr int C_   = 128;
constexpr int N_   = 4096;
constexpr int NB_  = 4;
constexpr int M_   = 1024;
constexpr int KNN_ = 32;
constexpr int QO_  = 32;
constexpr int VO_  = 128;
constexpr int NT_  = N_ + NB_;   // 4100
constexpr int NTP_ = 4128;       // NT padded to a multiple of 32

typedef __attribute__((ext_vector_type(16))) __bf16 v16bf;
typedef __attribute__((ext_vector_type(8)))  float  v8f;

#if defined(__gfx1250__) && __has_builtin(__builtin_amdgcn_tensor_load_to_lds)
#define USE_TDM 1
#else
#define USE_TDM 0
#endif

DI v8f wmma_bf16(v16bf a, v16bf b, v8f c) {
  // 8 args: (neg_a, A, neg_b, B, c_mod, C, reuse_a, reuse_b)
  return __builtin_amdgcn_wmma_f32_16x16x32_bf16(false, a, false, b, (short)0, c,
                                                 false, false);
}

// Compensated product: (Ah+Al)*(Bh+Bl) ~= Al*Bh + Ah*Bl + Ah*Bh  (drop Al*Bl)
DI v8f wmma_comp(v16bf ah, v16bf al, v16bf bh, v16bf bl, v8f c) {
  c = wmma_bf16(al, bh, c);
  c = wmma_bf16(ah, bl, c);
  c = wmma_bf16(ah, bh, c);
  return c;
}

// A-matrix fragment: 16x32 bf16, row-major source with leading dim `lda`.
// Layout: lane<16 -> M=lane, VGPR v: K = 2v;
// lane>=16 adds +16 to the K base. Each VGPR packs K (lo16) and K+1 (hi16).
DI v16bf load_a_frag(const __bf16* A, int lda) {
  int lane = threadIdx.x & 31;
  int m    = lane & 15;
  int hi   = (lane >> 4) & 1;
  v16bf f;
#pragma unroll
  for (int v = 0; v < 8; ++v) {
    int kb = 2 * v + (hi ? 16 : 0);
    f[2 * v]     = A[(size_t)m * lda + kb];
    f[2 * v + 1] = A[(size_t)m * lda + kb + 1];
  }
  return f;
}

// B-matrix fragment: 32x16 bf16 where the source is stored as rows over the
// *N* dimension, i.e. element B[k][col] = P[col*ldb + k] (P is 16 rows x 32).
DI v16bf load_b_frag_nk(const __bf16* P, int ldb) {
  int lane = threadIdx.x & 31;
  int col  = lane & 15;
  int hi   = (lane >> 4) & 1;
  v16bf f;
#pragma unroll
  for (int v = 0; v < 8; ++v) {
    int kb = 2 * v + (hi ? 16 : 0);
    f[2 * v]     = P[(size_t)col * ldb + kb];
    f[2 * v + 1] = P[(size_t)col * ldb + kb + 1];
  }
  return f;
}

// B-matrix fragment where source is stored [k][col] row-major with ld `ldb`.
DI v16bf load_b_frag_kn(const __bf16* P, int ldb) {
  int lane = threadIdx.x & 31;
  int col  = lane & 15;
  int hi   = (lane >> 4) & 1;
  v16bf f;
#pragma unroll
  for (int v = 0; v < 8; ++v) {
    int kb = 2 * v + (hi ? 16 : 0);
    f[2 * v]     = P[(size_t)kb * ldb + col];
    f[2 * v + 1] = P[(size_t)(kb + 1) * ldb + col];
  }
  return f;
}

DI unsigned fenc(float f) {  // order-preserving float -> unsigned encoding
  unsigned u = __float_as_uint(f);
  return u ^ ((u >> 31) ? 0xFFFFFFFFu : 0x80000000u);
}
DI float fdec(unsigned u) {
  unsigned b = u ^ ((u >> 31) ? 0x80000000u : 0xFFFFFFFFu);
  return __uint_as_float(b);
}
DI void bfsplit(float x, __bf16* h, __bf16* l) {
  __bf16 hh = (__bf16)x;
  *h = hh;
  *l = (__bf16)(x - (float)hh);
}

#if USE_TDM
typedef __attribute__((ext_vector_type(4))) unsigned v4u_;
typedef __attribute__((ext_vector_type(8))) int      v8i_;
typedef __attribute__((ext_vector_type(4))) int      v4i_;

// Build a 2D-tile D# per CDNA5 ISA and issue
// TENSOR_LOAD_TO_LDS (clang-23 6-arg form: g0, g1, g2, g3, g4, cpol).
// data_size = 2 bytes (bf16). tensor: tensor_d0 x tensor_d1 elements with
// row stride stride0 elements; tile: tile_d0 x tile_d1 elements starting at
// gptr; destination: lds_off (bytes).
DI void tdm_load_tile_2d(unsigned lds_off, const void* gptr,
                         unsigned tensor_d0, unsigned tensor_d1,
                         unsigned tile_d0, unsigned tile_d1,
                         unsigned stride0) {
  unsigned long long ga = (unsigned long long)(uintptr_t)gptr;
  v4u_ g0;
  g0[0] = 1u;                        // count=1, user descriptor, no gather
  g0[1] = lds_off;                   // LDS byte address
  g0[2] = (unsigned)ga;              // global_addr[31:0]
  g0[3] = (unsigned)((ga >> 32) & 0x01FFFFFFull) | (2u << 30);  // [56:32]|type=2
  v8i_ g1;
  g1[0] = (int)(1u << 16);           // workgroup_mask=0, data_size=1 (2B)
  g1[1] = (int)((tensor_d0 & 0xFFFFu) << 16);                   // tdim0 lo16
  g1[2] = (int)(((tensor_d0 >> 16) & 0xFFFFu) |
                ((tensor_d1 & 0xFFFFu) << 16));                 // tdim0 hi|tdim1 lo
  g1[3] = (int)(((tensor_d1 >> 16) & 0xFFFFu) |
                ((tile_d0 & 0xFFFFu) << 16));                   // tdim1 hi|tile0
  g1[4] = (int)(tile_d1 & 0xFFFFu);  // tile_dim1 (tile_dim2 = 0)
  g1[5] = (int)stride0;              // tensor_dim0_stride[31:0]
  g1[6] = 0;                         // stride0 hi16, stride1 lo16
  g1[7] = 0;                         // stride1 hi
  v4i_ gz4 = {0, 0, 0, 0};
  v8i_ gz8 = {0, 0, 0, 0, 0, 0, 0, 0};
  __builtin_amdgcn_tensor_load_to_lds(g0, g1, gz4, gz4, gz8, 0);
}
#endif

// ---------------------------------------------------------------------------
__global__ void zero_u32(unsigned* p, size_t n) {
  size_t i = (size_t)blockIdx.x * blockDim.x + threadIdx.x;
  size_t s = (size_t)gridDim.x * blockDim.x;
  for (; i < n; i += s) p[i] = 0u;
}

// ---------------------------------------------------------------------------
// K1: projections. One thread per output column n (coalesced x reads).
__global__ __launch_bounds__(64) void proj_kernel(
    const float* __restrict__ x, const float* __restrict__ tok,
    const float* __restrict__ Wq, const float* __restrict__ Wk,
    const float* __restrict__ Wv, float* __restrict__ sq,
    __bf16* __restrict__ xh, __bf16* __restrict__ xl, float* __restrict__ qf,
    __bf16* __restrict__ qh, __bf16* __restrict__ ql, float* __restrict__ kf,
    __bf16* __restrict__ kh, __bf16* __restrict__ kl, __bf16* __restrict__ vh,
    __bf16* __restrict__ vl) {
  constexpr int NBLK = NTP_ / 64;
  int blk = blockIdx.x;
  int b   = blk / (NBLK + 1);
  int nb  = blk % (NBLK + 1);
  int tid = threadIdx.x;
  int n   = nb * 64 + tid;
  __shared__ float X[C_ * 64];
  if (n < NTP_) {
    for (int c = 0; c < C_; ++c) {
      float v = 0.0f;
      if (n < N_) v = x[((size_t)b * C_ + c) * N_ + n];
      else if (n < NT_) v = tok[c * NB_ + (n - N_)];
      X[c * 64 + tid] = v;
    }
  }
  if (n < NTP_) {
    for (int o = 0; o < QO_; ++o) {
      float acc = 0.0f;
      for (int c = 0; c < C_; ++c) acc += Wk[o * C_ + c] * X[c * 64 + tid];
      size_t idx = (size_t)(b * NTP_ + n) * QO_ + o;
      kf[idx] = acc;
      bfsplit(acc, &kh[idx], &kl[idx]);
    }
  }
  if (n < N_) {
    float s = 0.0f;
    for (int c = 0; c < C_; ++c) {
      float v = X[c * 64 + tid];
      s += v * v;
      size_t xi = (size_t)(b * N_ + n) * C_ + c;
      bfsplit(v, &xh[xi], &xl[xi]);
    }
    sq[b * N_ + n] = s;
    for (int o = 0; o < QO_; ++o) {
      float acc = 0.0f;
      for (int c = 0; c < C_; ++c) acc += Wq[o * C_ + c] * X[c * 64 + tid];
      size_t idx = (size_t)(b * N_ + n) * QO_ + o;
      qf[idx] = acc;
      bfsplit(acc, &qh[idx], &ql[idx]);
    }
    for (int o = 0; o < VO_; ++o) {
      float acc = 0.0f;
      for (int c = 0; c < C_; ++c) acc += Wv[o * C_ + c] * X[c * 64 + tid];
      size_t vi = (size_t)(b * N_ + n) * VO_ + o;
      bfsplit(acc, &vh[vi], &vl[vi]);
    }
  }
}

// ---------------------------------------------------------------------------
// K2: Gram matrix (X^T X) via compensated bf16 WMMA + per-row top-32 select.
__global__ __launch_bounds__(64) void gram_knn_kernel(
    const __bf16* __restrict__ xh, const __bf16* __restrict__ xl,
    const float* __restrict__ sq, int* __restrict__ nn) {
  int blk = blockIdx.x;
  int b   = blk / (N_ / 16);
  int nb  = blk % (N_ / 16);
  int tid = threadIdx.x, lane = tid & 31, w = tid >> 5;
  __shared__ float DT[16 * 32];
  __shared__ float SQQ[16];
  const __bf16* xhb = xh + (size_t)b * N_ * C_;
  const __bf16* xlb = xl + (size_t)b * N_ * C_;
  if (tid < 16) SQQ[tid] = sq[b * N_ + nb * 16 + tid];
  v16bf aH[4], aL[4];
#pragma unroll
  for (int kc = 0; kc < 4; ++kc) {
    aH[kc] = load_a_frag(xhb + (size_t)(nb * 16) * C_ + kc * 32, C_);
    aL[kc] = load_a_frag(xlb + (size_t)(nb * 16) * C_ + kc * 32, C_);
  }
  float bd[KNN_];
  int   bi[KNN_];
#pragma unroll
  for (int i = 0; i < KNN_; ++i) { bd[i] = __builtin_inff(); bi[i] = 0; }
  __syncthreads();
  for (int mt = 0; mt < N_ / 32; ++mt) {
    int mcol0 = mt * 32 + w * 16;
    __builtin_prefetch(xhb + (size_t)(mcol0 + 32) * C_, 0, 1);
    v8f acc = {};
#pragma unroll
    for (int kc = 0; kc < 4; ++kc) {
      v16bf bh = load_b_frag_nk(xhb + (size_t)mcol0 * C_ + kc * 32, C_);
      v16bf bl = load_b_frag_nk(xlb + (size_t)mcol0 * C_ + kc * 32, C_);
      acc = wmma_comp(aH[kc], aL[kc], bh, bl, acc);
    }
    int colL = lane & 15, rb = (lane >> 4) * 8;
    float sqc = sq[b * N_ + mcol0 + colL];
#pragma unroll
    for (int r = 0; r < 8; ++r)
      DT[(rb + r) * 32 + w * 16 + colL] = SQQ[rb + r] - 2.0f * acc[r] + sqc;
    __syncthreads();
    if (tid < 16) {
      for (int j = 0; j < 32; ++j) {
        float d = DT[tid * 32 + j];
        if (d < bd[KNN_ - 1]) {
          int p = KNN_ - 1;
          while (p > 0 && bd[p - 1] > d) {
            bd[p] = bd[p - 1]; bi[p] = bi[p - 1]; --p;
          }
          bd[p] = d; bi[p] = mt * 32 + j;
        }
      }
    }
    __syncthreads();
  }
  if (tid < 16) {
    int base = (b * N_ + nb * 16 + tid) * KNN_;
    for (int i = 0; i < KNN_; ++i) nn[base + i] = bi[i];
  }
}

// ---------------------------------------------------------------------------
// K3: energy rows via WMMA, online softmax stats, bin maxima (atomicMax).
__global__ __launch_bounds__(64) void energy_kernel(
    const __bf16* __restrict__ qh, const __bf16* __restrict__ ql,
    const __bf16* __restrict__ kh, const __bf16* __restrict__ kl,
    float* __restrict__ rowmax, float* __restrict__ rowsum,
    unsigned* __restrict__ binmaxU) {
  int blk = blockIdx.x;
  int b   = blk / (N_ / 16);
  int nb  = blk % (N_ / 16);
  int tid = threadIdx.x, lane = tid & 31, w = tid >> 5;
  __shared__ float ET[16 * 32];
  v16bf aH = load_a_frag(qh + (size_t)(b * N_ + nb * 16) * QO_, QO_);
  v16bf aL = load_a_frag(ql + (size_t)(b * N_ + nb * 16) * QO_, QO_);
  float mx = -__builtin_inff(), sm = 0.0f;
  for (int mt = 0; mt < NTP_ / 32; ++mt) {
    int m0 = mt * 32 + w * 16;
    v16bf bh = load_b_frag_nk(kh + (size_t)(b * NTP_ + m0) * QO_, QO_);
    v16bf bl = load_b_frag_nk(kl + (size_t)(b * NTP_ + m0) * QO_, QO_);
    v8f acc = {};
    acc = wmma_comp(aH, aL, bh, bl, acc);
    int colL = lane & 15, rb = (lane >> 4) * 8;
#pragma unroll
    for (int r = 0; r < 8; ++r) ET[(rb + r) * 32 + w * 16 + colL] = acc[r];
    __syncthreads();
    if (tid < 16) {
      float tmx = mx;
      for (int j = 0; j < 32; ++j) {
        int m = mt * 32 + j;
        if (m < N_) tmx = fmaxf(tmx, ET[tid * 32 + j]);
      }
      if (tmx > mx) { sm *= __expf(mx - tmx); mx = tmx; }
      for (int j = 0; j < 32; ++j) {
        int m   = mt * 32 + j;
        float e = ET[tid * 32 + j];
        if (m < N_) sm += __expf(e - mx);
        else if (m < NT_) atomicMax(&binmaxU[b * NB_ + (m - N_)], fenc(e));
      }
    }
    __syncthreads();
  }
  if (tid < 16) {
    rowmax[b * N_ + nb * 16 + tid] = mx;
    rowsum[b * N_ + nb * 16 + tid] = sm;
  }
}

// ---------------------------------------------------------------------------
// K4: scatter attention mass of each row's 32 neighbors (deterministic:
// fixed-point u64 atomics) + neighbor counts.
__global__ __launch_bounds__(256) void score_kernel(
    const float* __restrict__ qf, const float* __restrict__ kf,
    const int* __restrict__ nn, const float* __restrict__ rowmax,
    const float* __restrict__ rowsum, unsigned long long* __restrict__ scnum,
    unsigned* __restrict__ scnt) {
  int t = blockIdx.x * blockDim.x + threadIdx.x;
  if (t >= B_ * N_) return;
  int b = t / N_;
  float qreg[QO_];
  const float* qr = qf + (size_t)t * QO_;
#pragma unroll
  for (int o = 0; o < QO_; ++o) qreg[o] = qr[o];
  float mx = rowmax[t], sm = rowsum[t];
  for (int k = 0; k < KNN_; ++k) {
    int idx = nn[(size_t)t * KNN_ + k];
    const float* kr = kf + (size_t)(b * NTP_ + idx) * QO_;
    float e = 0.0f;
#pragma unroll
    for (int o = 0; o < QO_; ++o) e += qreg[o] * kr[o];
    float a = __expf(e - mx) / sm;  // attn value in [0,1]
    unsigned long long fx = (unsigned long long)((double)a * 4294967296.0);
    atomicAdd(&scnum[b * N_ + idx], fx);
    atomicAdd(&scnt[b * N_ + idx], 1u);
  }
}

__global__ __launch_bounds__(256) void finalize_kernel(
    const unsigned long long* __restrict__ scnum,
    const unsigned* __restrict__ scnt, float* __restrict__ score) {
  int t = blockIdx.x * blockDim.x + threadIdx.x;
  if (t >= B_ * N_) return;
  double num = (double)scnum[t] / 4294967296.0;
  double den = (double)scnt[t] + 1e-8;
  score[t]   = (float)(num / (den * den));
}

// ---------------------------------------------------------------------------
// K5: per-batch bitonic argsort (descending score, ties -> lower index first).
__global__ __launch_bounds__(1024) void sort_kernel(
    const float* __restrict__ score, int* __restrict__ order) {
  int b = blockIdx.x;
  __shared__ float sk[N_];
  __shared__ int   sv[N_];
  for (int i = threadIdx.x; i < N_; i += 1024) {
    sk[i] = score[b * N_ + i];
    sv[i] = i;
  }
  __syncthreads();
  for (int k = 2; k <= N_; k <<= 1) {
    for (int j = k >> 1; j > 0; j >>= 1) {
      for (int i = threadIdx.x; i < N_; i += 1024) {
        int ixj = i ^ j;
        if (ixj > i) {
          bool  up = ((i & k) == 0);
          float a = sk[i], c = sk[ixj];
          int   ia = sv[i], ic = sv[ixj];
          bool aFirst = (a > c) || (a == c && ia < ic);  // descending
          bool doSwap = up ? !aFirst : aFirst;
          if (doSwap) { sk[i] = c; sk[ixj] = a; sv[i] = ic; sv[ixj] = ia; }
        }
      }
      __syncthreads();
    }
  }
  for (int i = threadIdx.x; i < N_; i += 1024) order[b * N_ + i] = sv[i];
}

// ---------------------------------------------------------------------------
// K6: bin softmax -> k_bins -> searchsorted -> idx_down.
__global__ __launch_bounds__(1024) void bins_kernel(
    const unsigned* __restrict__ binmaxU, const int* __restrict__ order,
    int* __restrict__ idxd, float* __restrict__ out_idx) {
  int b = blockIdx.x;
  __shared__ int cum[NB_], prev[NB_];
  if (threadIdx.x == 0) {
    float e[NB_], mx = -__builtin_inff();
    for (int j = 0; j < NB_; ++j) {
      e[j] = fdec(binmaxU[b * NB_ + j]);
      mx   = fmaxf(mx, e[j]);
    }
    float s = 0.0f;
    for (int j = 0; j < NB_; ++j) { e[j] = __expf(e[j] - mx); s += e[j]; }
    int kb[NB_], tot = 0;
    for (int j = 0; j < NB_ - 1; ++j) {
      kb[j] = (int)floorf(2.0f * (float)M_ / (float)NB_ * (e[j] / s));
      tot += kb[j];
    }
    kb[NB_ - 1] = M_ - tot;
    int c = 0;
    for (int j = 0; j < NB_; ++j) { prev[j] = c; c += kb[j]; cum[j] = c; }
  }
  __syncthreads();
  for (int s = threadIdx.x; s < M_; s += blockDim.x) {
    int bin = 0;
    while (bin < NB_ - 1 && s >= cum[bin]) ++bin;  // searchsorted 'right'
    int off  = s - prev[bin];
    int flat = bin * (N_ / NB_) + off;
    int idx  = order[b * N_ + flat];
    idxd[b * M_ + s]    = idx;
    out_idx[b * M_ + s] = (float)idx;
  }
}

// ---------------------------------------------------------------------------
// K7: recompute attention rows for the selected sources; store bf16 hi/lo.
__global__ __launch_bounds__(64) void attn_down_kernel(
    const __bf16* __restrict__ qh, const __bf16* __restrict__ ql,
    const __bf16* __restrict__ kh, const __bf16* __restrict__ kl,
    const int* __restrict__ idxd, const float* __restrict__ rowmax,
    const float* __restrict__ rowsum, __bf16* __restrict__ adh,
    __bf16* __restrict__ adl) {
  int blk = blockIdx.x;
  int b   = blk / (M_ / 16);
  int mb  = blk % (M_ / 16);
  int tid = threadIdx.x, lane = tid & 31, w = tid >> 5;
  __shared__ __bf16 AQH[16 * QO_], AQL[16 * QO_];
  __shared__ float  RMX[16], RSM[16];
  if (tid < 16) {
    int src  = idxd[b * M_ + mb * 16 + tid];
    RMX[tid] = rowmax[b * N_ + src];
    RSM[tid] = rowsum[b * N_ + src];
    for (int o = 0; o < QO_; ++o) {
      AQH[tid * QO_ + o] = qh[(size_t)(b * N_ + src) * QO_ + o];
      AQL[tid * QO_ + o] = ql[(size_t)(b * N_ + src) * QO_ + o];
    }
  }
  __syncthreads();
  v16bf aH = load_a_frag(AQH, QO_);
  v16bf aL = load_a_frag(AQL, QO_);
  for (int nt = 0; nt < N_ / 32; ++nt) {
    int n0 = nt * 32 + w * 16;
    v16bf bh = load_b_frag_nk(kh + (size_t)(b * NTP_ + n0) * QO_, QO_);
    v16bf bl = load_b_frag_nk(kl + (size_t)(b * NTP_ + n0) * QO_, QO_);
    v8f acc = {};
    acc = wmma_comp(aH, aL, bh, bl, acc);
    int colL = lane & 15, rb = (lane >> 4) * 8;
#pragma unroll
    for (int r = 0; r < 8; ++r) {
      int   mr = rb + r;
      float a  = __expf(acc[r] - RMX[mr]) / RSM[mr];
      size_t o = (size_t)(b * M_ + mb * 16 + mr) * N_ + n0 + colL;
      bfsplit(a, &adh[o], &adl[o]);
    }
  }
}

// ---------------------------------------------------------------------------
// K8: x_ds = attn_down (MxN) @ vT (NxVO), compensated bf16 WMMA.
// B tiles (32 K-rows x 32 o-cols of vT, shared by both waves) are staged into
// LDS by the Tensor Data Mover, double-buffered: the TDM fetch of chunk nt+1
// overlaps the WMMAs on chunk nt. Wave 0 issues the DMA (TENSORcnt is
// per-wave) and the workgroup barrier publishes the LDS tile to wave 1.
__global__ __launch_bounds__(64) void xds_kernel(
    const __bf16* __restrict__ adh, const __bf16* __restrict__ adl,
    const __bf16* __restrict__ vh, const __bf16* __restrict__ vl,
    float* __restrict__ out) {
  int blk = blockIdx.x;
  int ob  = blk & 3;          // VO/32 = 4
  int mb  = (blk >> 2) & 63;  // M/16 = 64
  int b   = blk >> 8;
  int lane = threadIdx.x & 31, w = threadIdx.x >> 5;
  const __bf16* ah  = adh + (size_t)(b * M_ + mb * 16) * N_;
  const __bf16* al  = adl + (size_t)(b * M_ + mb * 16) * N_;
  const __bf16* vhb = vh + (size_t)b * N_ * VO_ + ob * 32;
  const __bf16* vlb = vl + (size_t)b * N_ * VO_ + ob * 32;
  __shared__ __bf16 BTH[2][32 * 32];
  __shared__ __bf16 BTL[2][32 * 32];
  v8f acc = {};
#if USE_TDM
  if (threadIdx.x < 32) {  // wave 0 only: prime buffer 0
    tdm_load_tile_2d((unsigned)(uintptr_t)&BTH[0][0], vhb, VO_, N_, 32, 32, VO_);
    tdm_load_tile_2d((unsigned)(uintptr_t)&BTL[0][0], vlb, VO_, N_, 32, 32, VO_);
    __builtin_amdgcn_s_wait_tensorcnt(0);
  }
  __syncthreads();
#endif
  for (int nt = 0; nt < N_ / 32; ++nt) {
    int cur = nt & 1;
#if USE_TDM
    if (threadIdx.x < 32 && nt + 1 < N_ / 32) {  // prefetch next chunk via TDM
      tdm_load_tile_2d((unsigned)(uintptr_t)&BTH[cur ^ 1][0],
                       vhb + (size_t)(nt + 1) * 32 * VO_, VO_, N_, 32, 32, VO_);
      tdm_load_tile_2d((unsigned)(uintptr_t)&BTL[cur ^ 1][0],
                       vlb + (size_t)(nt + 1) * 32 * VO_, VO_, N_, 32, 32, VO_);
    }
#endif
    __builtin_prefetch(ah + (size_t)(nt + 1) * 32, 0, 1);
    v16bf Ah = load_a_frag(ah + nt * 32, N_);
    v16bf Al = load_a_frag(al + nt * 32, N_);
#if USE_TDM
    v16bf Bh = load_b_frag_kn(&BTH[cur][w * 16], 32);
    v16bf Bl = load_b_frag_kn(&BTL[cur][w * 16], 32);
#else
    v16bf Bh = load_b_frag_kn(vhb + (size_t)nt * 32 * VO_ + w * 16, VO_);
    v16bf Bl = load_b_frag_kn(vlb + (size_t)nt * 32 * VO_ + w * 16, VO_);
#endif
    acc = wmma_comp(Ah, Al, Bh, Bl, acc);
#if USE_TDM
    if (threadIdx.x < 32) __builtin_amdgcn_s_wait_tensorcnt(0);
    __syncthreads();
#endif
  }
  int colL = lane & 15, rb = (lane >> 4) * 8;
  int o = ob * 32 + w * 16 + colL;
#pragma unroll
  for (int r = 0; r < 8; ++r) {
    int m = mb * 16 + rb + r;
    out[((size_t)b * VO_ + o) * M_ + m] = acc[r];
  }
}

// ---------------------------------------------------------------------------
extern "C" void kernel_launch(void* const* d_in, const int* in_sizes, int n_in,
                              void* d_out, int out_size, void* d_ws,
                              size_t ws_size, hipStream_t stream) {
  (void)in_sizes; (void)n_in; (void)out_size; (void)ws_size;
  const float* x   = (const float*)d_in[0];
  const float* tok = (const float*)d_in[1];
  const float* Wq  = (const float*)d_in[2];
  const float* Wk  = (const float*)d_in[3];
  const float* Wv  = (const float*)d_in[4];
  float*       out = (float*)d_out;

  char*  w   = (char*)d_ws;
  size_t off = 0;
  auto alloc = [&](size_t bytes) -> void* {
    void* p = w + off;
    off = (off + bytes + 255) & ~(size_t)255;
    return p;
  };

  __bf16* xh = (__bf16*)alloc((size_t)B_ * N_ * C_ * 2);
  __bf16* xl = (__bf16*)alloc((size_t)B_ * N_ * C_ * 2);
  float*  qf = (float*)alloc((size_t)B_ * N_ * QO_ * 4);
  __bf16* qh = (__bf16*)alloc((size_t)B_ * N_ * QO_ * 2);
  __bf16* ql = (__bf16*)alloc((size_t)B_ * N_ * QO_ * 2);
  float*  kf = (float*)alloc((size_t)B_ * NTP_ * QO_ * 4);
  __bf16* kh = (__bf16*)alloc((size_t)B_ * NTP_ * QO_ * 2);
  __bf16* kl = (__bf16*)alloc((size_t)B_ * NTP_ * QO_ * 2);
  __bf16* vh = (__bf16*)alloc((size_t)B_ * N_ * VO_ * 2);
  __bf16* vl = (__bf16*)alloc((size_t)B_ * N_ * VO_ * 2);
  float*  sq = (float*)alloc((size_t)B_ * N_ * 4);
  float*  rowmax = (float*)alloc((size_t)B_ * N_ * 4);
  float*  rowsum = (float*)alloc((size_t)B_ * N_ * 4);
  int*    nn     = (int*)alloc((size_t)B_ * N_ * KNN_ * 4);
  unsigned* binmax = (unsigned*)alloc((size_t)B_ * NB_ * 4);
  unsigned long long* scnum =
      (unsigned long long*)alloc((size_t)B_ * N_ * 8);
  unsigned* scnt  = (unsigned*)alloc((size_t)B_ * N_ * 4);
  float*    score = (float*)alloc((size_t)B_ * N_ * 4);
  int*      order = (int*)alloc((size_t)B_ * N_ * 4);
  int*      idxd  = (int*)alloc((size_t)B_ * M_ * 4);
  __bf16*   adh   = (__bf16*)alloc((size_t)B_ * M_ * N_ * 2);
  __bf16*   adl   = (__bf16*)alloc((size_t)B_ * M_ * N_ * 2);

  zero_u32<<<256, 256, 0, stream>>>((unsigned*)scnum, (size_t)B_ * N_ * 2);
  zero_u32<<<64, 256, 0, stream>>>(scnt, (size_t)B_ * N_);
  zero_u32<<<1, 64, 0, stream>>>(binmax, (size_t)B_ * NB_);

  proj_kernel<<<B_ * (NTP_ / 64 + 1), 64, 0, stream>>>(
      x, tok, Wq, Wk, Wv, sq, xh, xl, qf, qh, ql, kf, kh, kl, vh, vl);
  gram_knn_kernel<<<B_ * (N_ / 16), 64, 0, stream>>>(xh, xl, sq, nn);
  energy_kernel<<<B_ * (N_ / 16), 64, 0, stream>>>(qh, ql, kh, kl, rowmax,
                                                   rowsum, binmax);
  score_kernel<<<(B_ * N_ + 255) / 256, 256, 0, stream>>>(qf, kf, nn, rowmax,
                                                          rowsum, scnum, scnt);
  finalize_kernel<<<(B_ * N_ + 255) / 256, 256, 0, stream>>>(scnum, scnt,
                                                             score);
  sort_kernel<<<B_, 1024, 0, stream>>>(score, order);
  bins_kernel<<<B_, 1024, 0, stream>>>(binmax, order, idxd,
                                       out + (size_t)B_ * VO_ * M_);
  attn_down_kernel<<<B_ * (M_ / 16), 64, 0, stream>>>(
      qh, ql, kh, kl, idxd, rowmax, rowsum, adh, adl);
  xds_kernel<<<B_ * (M_ / 16) * (VO_ / 32), 64, 0, stream>>>(adh, adl, vh, vl,
                                                             out);
}
